// QuantLinear_54717883351703
// MI455X (gfx1250) — matrix operand order repaired and checked by reading, not compiled, not run on hardware
//
#include <hip/hip_runtime.h>

typedef _Float16 v16h __attribute__((ext_vector_type(16)));
typedef float    v8f  __attribute__((ext_vector_type(8)));
typedef _Float16 h8   __attribute__((ext_vector_type(8)));
typedef _Float16 h2   __attribute__((ext_vector_type(2)));

#define INF   4096
#define OUTF  4096
#define GROUP 128
#define BM    128
#define BN    128
#define BK    64                 // K staged per barrier = 2 WMMA k-steps
#define NKS   (INF / BK)         // 64 stages

// NOTE on K ordering: within every 8 consecutive original K values we store the
// permuted order (0,4),(1,5),(2,6),(3,7) in LDS -- identically for A and B, so
// the contraction is unchanged. This makes the packed nibble dequant free.

// ---- fragment gathers from LDS (CDNA5 WMMA VGPR layouts, stored-K space) ----

// A-matrix 16x32 f16: lanes 0-15 -> M=lane, K pairs {0..7 | 16..23} in v0..v7
//                     lanes 16-31 -> same M, K pairs {8..15 | 24..31}
__device__ __forceinline__ v16h load_frag_a(const _Float16* rowBase, int half) {
    union { v16h v; unsigned u[8]; } a;
    const unsigned* p = (const unsigned*)rowBase;   // word w -> stored K=2w,2w+1
#pragma unroll
    for (int v = 0; v < 8; ++v) {
        int w = ((v >> 2) << 3) + (half << 2) + (v & 3);
        a.u[v] = p[w];          // two contiguous 16B runs -> 2x ds_load_b128
    }
    return a.v;
}

// B-matrix 32x16 f16 stored K-contiguous per column: Bs[col][k].
// lanes 0-15 -> K pairs {0..7 | 16..23}, lanes 16-31 -> K pairs {8..15 | 24..31}, col = lane&15
__device__ __forceinline__ v16h load_frag_b(const _Float16* colBase, int half) {
    union { v16h v; unsigned u[8]; } b;
    const unsigned* p = (const unsigned*)colBase;
#pragma unroll
    for (int v = 0; v < 8; ++v)
        b.u[v] = p[((v >> 2) << 3) + (half << 2) + (v & 3)];  // two contiguous 16B runs -> 2x ds_load_b128
    return b.v;
}

__global__ __launch_bounds__(256)
void qlinear_wmma_f16(const float* __restrict__ x,        // [M, INF] fp32
                      const int*   __restrict__ qweight,  // [INF/8, OUTF] int32
                      const int*   __restrict__ qzeros,   // [INF/GROUP, OUTF/8] int32
                      const float* __restrict__ scales,   // [INF/GROUP, OUTF] fp32
                      const float* __restrict__ bias,     // [OUTF]
                      float* __restrict__ out)            // [M, OUTF]
{
    // ping-pong double buffers: 2 * (16 KB A + 16 KB B) = 64 KB
    __shared__ _Float16 As[2][BM][BK];
    __shared__ _Float16 Bs[2][BN][BK];   // transposed [n][k], K contiguous

    const int tid  = threadIdx.x;
    const int lane = tid & 31;
    const int wave = tid >> 5;                  // 0..7
    const int half = lane >> 4;
    const int l    = lane & 15;

    const int colBase = blockIdx.x * BN;
    const int rowBase = blockIdx.y * BM;

    // 2 (M) x 4 (N) wave grid; wave owns 64x32 = 4x2 WMMA tiles
    const int waveM = (wave & 1) * 64;
    const int waveN = (wave >> 1) * 32;

    v8f acc[4][2];
#pragma unroll
    for (int t = 0; t < 4; ++t)
#pragma unroll
        for (int u = 0; u < 2; ++u)
            acc[t][u] = v8f{};

    // staging assignments
    const int aRow  = tid >> 1;                 // 0..127, 2 threads/row
    const int aKoff = (tid & 1) * 32;           // 32 floats each
    const int bCol  = tid & 127;                // rep-invariant column
    const int bR0   = tid >> 7;                 // 0..1 -> packed rows {bR0, bR0+2, ..}

    auto stage = [&](int k0, int buf) {
        // ---- A: 128 x 64 fp32 -> f16, permuted {0,4,1,5,2,6,3,7} per 8-group ----
        const float4* xv =
            (const float4*)(x + (size_t)(rowBase + aRow) * INF + k0 + aKoff);
#pragma unroll
        for (int i = 0; i < 4; ++i) {
            float4 f0 = xv[2 * i], f1 = xv[2 * i + 1];   // K = 0..3 | 4..7 of group
            h8 h;                                        // zero-cost permute in cvt_pk
            h[0]=(_Float16)f0.x; h[1]=(_Float16)f1.x;
            h[2]=(_Float16)f0.y; h[3]=(_Float16)f1.y;
            h[4]=(_Float16)f0.z; h[5]=(_Float16)f1.z;
            h[6]=(_Float16)f0.w; h[7]=(_Float16)f1.w;
            *(h8*)&As[buf][aRow][aKoff + i * 8] = h;
        }
        // ---- B: packed nibble dequant, 64(K) x 128(N) ----
        const int   g    = k0 / GROUP;           // 64-row slice never crosses a group
        const int   gcol = colBase + bCol;
        const int   z    = ((qzeros[(size_t)g * (OUTF / 8) + (gcol >> 3)]
                             >> ((gcol & 7) * 4)) & 15) + 1;
        const float s    = scales[(size_t)g * OUTF + gcol];
        const _Float16 sh = (_Float16)s;
        const _Float16 ch = (_Float16)(float)(1024 + z); // exact in f16
        const h2 s2 = {sh, sh};
        const h2 c2 = {ch, ch};
#pragma unroll
        for (int rep = 0; rep < 4; ++rep) {
            const int r = bR0 + rep * 2;         // 0..7 across the workgroup
            const unsigned uq =
                (unsigned)qweight[(size_t)(k0 / 8 + r) * OUTF + gcol];
            // 0x6400 | w  ==  f16(1024 + w), exact for w in [0,15]
            union { unsigned u; h2 h; } t0, t1, t2, t3;
            t0.u = ( uq        & 0x000F000Fu) | 0x64006400u;  // (w0, w4)
            t1.u = ((uq >>  4) & 0x000F000Fu) | 0x64006400u;  // (w1, w5)
            t2.u = ((uq >>  8) & 0x000F000Fu) | 0x64006400u;  // (w2, w6)
            t3.u = ((uq >> 12) & 0x000F000Fu) | 0x64006400u;  // (w3, w7)
            union { h8 v; h2 p[4]; } wv;
            wv.p[0] = (t0.h - c2) * s2;          // exact (w-z), one rounding in mul
            wv.p[1] = (t1.h - c2) * s2;          // -> v_pk_add_f16 + v_pk_mul_f16
            wv.p[2] = (t2.h - c2) * s2;
            wv.p[3] = (t3.h - c2) * s2;
            *(h8*)&Bs[buf][bCol][r * 8] = wv.v;  // same permuted K order as A
        }
    };

    stage(0, 0);
    __syncthreads();

    for (int kt = 0; kt < NKS; ++kt) {
        const int buf = kt & 1;
        if (kt + 1 < NKS)
            stage((kt + 1) * BK, buf ^ 1);       // overlaps with WMMAs below

#pragma unroll
        for (int sub = 0; sub < 2; ++sub) {      // two 16x16x32 k-steps per stage
            v16h aF[4], bF[2];
#pragma unroll
            for (int t = 0; t < 4; ++t)
                aF[t] = load_frag_a(&As[buf][waveM + t * 16 + l][sub * 32], half);
#pragma unroll
            for (int u = 0; u < 2; ++u)
                bF[u] = load_frag_b(&Bs[buf][waveN + u * 16 + l][sub * 32], half);

#pragma unroll
            for (int t = 0; t < 4; ++t)
#pragma unroll
                for (int u = 0; u < 2; ++u)
                    acc[t][u] = __builtin_amdgcn_wmma_f32_16x16x32_f16(
                        /*neg_a=*/false, aF[t], /*neg_b=*/false, bF[u],
                        /*c_mod=*/(short)0, acc[t][u],
                        /*reuse_a=*/false, /*reuse_b=*/false);
        }
        __syncthreads();                         // one barrier per 16 WMMAs
    }

    // ---- epilogue: bias add + fp32 store (C/D layout: row = r + 8*half, col = l) ----
#pragma unroll
    for (int u = 0; u < 2; ++u) {
        const int col = colBase + waveN + u * 16 + l;
        const float b = bias[col];
#pragma unroll
        for (int t = 0; t < 4; ++t) {
            const int rbase = rowBase + waveM + t * 16 + half * 8;
#pragma unroll
            for (int r = 0; r < 8; ++r)
                out[(size_t)(rbase + r) * OUTF + col] = acc[t][u][r] + b;
        }
    }
}

extern "C" void kernel_launch(void* const* d_in, const int* in_sizes, int n_in,
                              void* d_out, int out_size, void* d_ws, size_t ws_size,
                              hipStream_t stream) {
    const float* x   = (const float*)d_in[0];
    const int*   qw  = (const int*)d_in[1];
    const int*   qz  = (const int*)d_in[2];
    const float* sc  = (const float*)d_in[3];
    const float* bs  = (const float*)d_in[4];
    float*       out = (float*)d_out;

    const int M = in_sizes[0] / INF;            // 8*512 = 4096
    dim3 grid(OUTF / BN, M / BM);               // 32 x 32
    qlinear_wmma_f16<<<grid, 256, 0, stream>>>(x, qw, qz, sc, bs, out);
}
